// QRNNLayer_77318001262592
// MI455X (gfx1250) — matrix operand order repaired and checked by reading, not compiled, not run on hardware
//
#include <hip/hip_runtime.h>
#include <hip/hip_bf16.h>

// ---------------------------------------------------------------------------
// QRNN layer for MI455X (gfx1250, wave32, WMMA).
//   y = inp @ W^T + b ; z=tanh, f=sig, o=sig ; h_t = f z + (1-f) h ; out = sig(o)*h
// GEMM uses V_WMMA_F32_16X16X32_BF16 with a bf16-hi/lo (bf16x3) decomposition
// for near-fp32 accuracy at bf16 matrix-core rates.
// Wave tile 32x64 (2x4 WMMA tiles): 24 WMMAs per 24 16-byte loads per K-chunk.
// ---------------------------------------------------------------------------

typedef __attribute__((ext_vector_type(16))) __bf16 v16bf;
typedef __attribute__((ext_vector_type(8)))  float  v8f;

#define BB   16
#define SS   4096
#define DD   512
#define HH   512
#define N3   1536
#define MM   (BB * SS)          // 65536 rows
#define SCH  64                 // scan chunk length
#define NCH  (SS / SCH)         // 64 chunks
#define CTOT (BB * HH)          // 8192 channels
#define TSTEP (16 * DD)         // shorts per 16-row tile step (16 KB)

// ---------------------------------------------------------------------------
// f32 -> (bf16 hi, bf16 lo) split, round-to-nearest-even on both parts.
// ---------------------------------------------------------------------------
__device__ __forceinline__ void split_bf(float x, unsigned short& h, unsigned short& l) {
    unsigned u  = __float_as_uint(x);
    unsigned rh = u + 0x7FFFu + ((u >> 16) & 1u);
    h = (unsigned short)(rh >> 16);
    float    hf = __uint_as_float(((unsigned)h) << 16);
    float    rs = x - hf;
    unsigned v  = __float_as_uint(rs);
    unsigned rl = v + 0x7FFFu + ((v >> 16) & 1u);
    l = (unsigned short)(rl >> 16);
}

// Convert n4*4 floats into packed bf16 hi/lo arrays.
__global__ void qrnn_split(const float* __restrict__ src,
                           unsigned short* __restrict__ hi,
                           unsigned short* __restrict__ lo, int n4) {
    int t = blockIdx.x * blockDim.x + threadIdx.x;
    if (t >= n4) return;
    float4 x = reinterpret_cast<const float4*>(src)[t];
    unsigned short hs0, hs1, hs2, hs3, ls0, ls1, ls2, ls3;
    split_bf(x.x, hs0, ls0);
    split_bf(x.y, hs1, ls1);
    split_bf(x.z, hs2, ls2);
    split_bf(x.w, hs3, ls3);
    uint2 hp, lp;
    hp.x = (unsigned)hs0 | ((unsigned)hs1 << 16);
    hp.y = (unsigned)hs2 | ((unsigned)hs3 << 16);
    lp.x = (unsigned)ls0 | ((unsigned)ls1 << 16);
    lp.y = (unsigned)ls2 | ((unsigned)ls3 << 16);
    reinterpret_cast<uint2*>(hi)[t] = hp;
    reinterpret_cast<uint2*>(lo)[t] = lp;
}

// ---------------------------------------------------------------------------
// WMMA helpers.
// ---------------------------------------------------------------------------
__device__ __forceinline__ v8f wmma_bf16(v16bf a, v16bf b, v8f c) {
    // (neg_a, A, neg_b, B, c_mod, C, reuse_a, reuse_b)
    return __builtin_amdgcn_wmma_f32_16x16x32_bf16(false, a, false, b, (short)0, c,
                                                   false, false);
}

// Load 16 bf16 (as ushort) fragment from two 16-byte runs.
__device__ __forceinline__ v16bf load_frag16(const unsigned short* p0,
                                             const unsigned short* p1) {
    union { v16bf v; uint4 q[2]; } u;
    u.q[0] = *reinterpret_cast<const uint4*>(p0);
    u.q[1] = *reinterpret_cast<const uint4*>(p1);
    return u.v;
}

__device__ __forceinline__ float sigf(float x) { return 1.0f / (1.0f + __expf(-x)); }

// ---------------------------------------------------------------------------
// GEMM + bias + gate activations.
//   block: 256 thr = 8 waves (4 M x 2 N); wave tile 32x64; block tile 128x128.
//   grid:  (N3/128 = 12, MM/128 = 512)
// A fragment (16x32 bf16): lane M = lane&15; lanes 0-15 hold K 0-7 & 16-23,
// lanes 16-31 hold K 8-15 & 24-31 (ISA 7.12.2).
// B fragment (32x16 bf16): lane N = lane&15; same K order as A,
// lanes 0-15 K 0-7 & 16-23, lanes 16-31 K 8-15 & 24-31. W is stored (N,K) row-major, so
// these are two 16-byte reads -- no transpose needed.
// Tile steps within a wave are 16-KB immediate offsets from 4 base pointers.
// waves_per_eu(2) gives the RA headroom to keep all fragments live so the
// inner loop uses partial s_wait_loadcnt instead of full drains.
// ---------------------------------------------------------------------------
__global__ __launch_bounds__(256)
__attribute__((amdgpu_waves_per_eu(2)))
void qrnn_gemm_wmma(
    const unsigned short* __restrict__ Ahi, const unsigned short* __restrict__ Alo,
    const unsigned short* __restrict__ Whi, const unsigned short* __restrict__ Wlo,
    const float* __restrict__ bias,
    float* __restrict__ zg, float* __restrict__ fg, float* __restrict__ outp) {
    const int lane  = threadIdx.x & 31;
    const int wave  = threadIdx.x >> 5;
    const int wm    = wave & 3;           // 4 M positions
    const int wn    = wave >> 2;          // 2 N positions
    const int row0  = blockIdx.y * 128 + wm * 32;
    const int n0    = blockIdx.x * 128 + wn * 64;
    const int lcol  = lane & 15;
    const int lhalf = lane >> 4;

    const unsigned short* aH = Ahi + (size_t)(row0 + lcol) * DD;
    const unsigned short* aL = Alo + (size_t)(row0 + lcol) * DD;
    const unsigned short* bH = Whi + (size_t)(n0 + lcol) * DD;
    const unsigned short* bL = Wlo + (size_t)(n0 + lcol) * DD;

    const int kA = lhalf * 8;    // A: K 0-7 / 8-15 per lane half
    const int kB = lhalf * 8;    // B: K 0-7 / 8-15 per lane half

    v8f acc[2][4];
#pragma unroll
    for (int mt = 0; mt < 2; ++mt)
#pragma unroll
        for (int nt = 0; nt < 4; ++nt) acc[mt][nt] = (v8f){};

#pragma unroll 2
    for (int kc = 0; kc < DD; kc += 32) {
        const int ka = kc + kA;
        const int kb = kc + kB;
        v16bf fAh[2], fAl[2], fBh[4], fBl[4];
#pragma unroll
        for (int mt = 0; mt < 2; ++mt) {
            fAh[mt] = load_frag16(aH + mt * TSTEP + ka, aH + mt * TSTEP + ka + 16);
            fAl[mt] = load_frag16(aL + mt * TSTEP + ka, aL + mt * TSTEP + ka + 16);
        }
#pragma unroll
        for (int nt = 0; nt < 4; ++nt) {
            fBh[nt] = load_frag16(bH + nt * TSTEP + kb, bH + nt * TSTEP + kb + 16);
            fBl[nt] = load_frag16(bL + nt * TSTEP + kb, bL + nt * TSTEP + kb + 16);
        }
        // bf16x3: hi*hi + hi*lo + lo*hi into the same f32 accumulator.
        // nt-outer so each B fragment's consumers are adjacent (short lifetime).
#pragma unroll
        for (int nt = 0; nt < 4; ++nt) {
#pragma unroll
            for (int mt = 0; mt < 2; ++mt) {
                acc[mt][nt] = wmma_bf16(fAh[mt], fBh[nt], acc[mt][nt]);
                acc[mt][nt] = wmma_bf16(fAh[mt], fBl[nt], acc[mt][nt]);
                acc[mt][nt] = wmma_bf16(fAl[mt], fBh[nt], acc[mt][nt]);
            }
        }
    }

    // Epilogue: bias + activation. Gate is uniform across the whole block
    // (block spans 128 N-columns inside one 512-aligned gate region).
    const int gate = n0 >> 9;                     // 0: z(tanh) 1: f(sig) 2: o(sig)
    float* dst = (gate == 0) ? zg : ((gate == 1) ? fg : outp);

#pragma unroll
    for (int mt = 0; mt < 2; ++mt) {
#pragma unroll
        for (int nt = 0; nt < 4; ++nt) {
            const int   n  = n0 + nt * 16 + lcol;
            const int   hc = n & (HH - 1);
            const float bv = bias[n];
            v8f a = acc[mt][nt];
#pragma unroll
            for (int r = 0; r < 8; ++r) {
                // C/D layout: VGPR r, lanes 0-15 -> M=r, lanes 16-31 -> M=8+r
                const int m   = row0 + mt * 16 + lhalf * 8 + r;
                const float v = a[r] + bv;
                const float res = (gate == 0) ? tanhf(v) : sigf(v);
                dst[(size_t)m * HH + hc] = res;
            }
        }
    }
}

// ---------------------------------------------------------------------------
// Scan pass 1: per (channel, chunk) compute decay a = prod(1-f) and local h.
// tid -> hc (0..511), b (0..15), j (0..63); consecutive tids -> consecutive hc.
// ---------------------------------------------------------------------------
__global__ __launch_bounds__(256) void qrnn_scan_chunks(
    const float* __restrict__ zg, const float* __restrict__ fg,
    float* __restrict__ cA, float* __restrict__ cH) {
    const int tid = blockIdx.x * blockDim.x + threadIdx.x;
    const int hc  = tid & (HH - 1);
    const int r   = tid >> 9;
    const int b   = r & (BB - 1);
    const int j   = r >> 4;
    size_t idx = ((size_t)(b * SS + j * SCH)) * HH + hc;
    float a = 1.0f, h = 0.0f;
#pragma unroll 8
    for (int s = 0; s < SCH; ++s) {
        const float f  = fg[idx];
        const float z  = zg[idx];
        const float om = 1.0f - f;
        h = f * z + om * h;
        a *= om;
        idx += HH;
    }
    const int c = b * HH + hc;
    cA[(size_t)j * CTOT + c] = a;
    cH[(size_t)j * CTOT + c] = h;
}

// ---------------------------------------------------------------------------
// Scan pass 2: combine chunk summaries along S; emit per-chunk carry-in and
// the final state c[:, -1].
// ---------------------------------------------------------------------------
__global__ __launch_bounds__(256) void qrnn_scan_combine(
    const float* __restrict__ cA, const float* __restrict__ cH,
    float* __restrict__ carry, float* __restrict__ clast) {
    const int c = blockIdx.x * blockDim.x + threadIdx.x;  // 0..8191 == b*H + h
    float h = 0.0f;
#pragma unroll 4
    for (int j = 0; j < NCH; ++j) {
        carry[(size_t)j * CTOT + c] = h;
        h = cH[(size_t)j * CTOT + c] + cA[(size_t)j * CTOT + c] * h;
    }
    clast[c] = h;
}

// ---------------------------------------------------------------------------
// Scan pass 3: replay each chunk with its carry-in; out holds sigmoid(o)
// (written by the GEMM epilogue) and is multiplied in place by c_t.
// ---------------------------------------------------------------------------
__global__ __launch_bounds__(256) void qrnn_scan_apply(
    const float* __restrict__ zg, const float* __restrict__ fg,
    const float* __restrict__ carry, float* __restrict__ out) {
    const int tid = blockIdx.x * blockDim.x + threadIdx.x;
    const int hc  = tid & (HH - 1);
    const int r   = tid >> 9;
    const int b   = r & (BB - 1);
    const int j   = r >> 4;
    const int c   = b * HH + hc;
    float h = carry[(size_t)j * CTOT + c];
    size_t idx = ((size_t)(b * SS + j * SCH)) * HH + hc;
#pragma unroll 8
    for (int s = 0; s < SCH; ++s) {
        const float f = fg[idx];
        const float z = zg[idx];
        h = f * z + (1.0f - f) * h;
        out[idx] = out[idx] * h;   // out held sigmoid(o)
        idx += HH;
    }
}

// ---------------------------------------------------------------------------
// Launcher.
// ---------------------------------------------------------------------------
extern "C" void kernel_launch(void* const* d_in, const int* in_sizes, int n_in,
                              void* d_out, int out_size, void* d_ws, size_t ws_size,
                              hipStream_t stream) {
    const float* inp  = (const float*)d_in[0];   // (16, 4096, 512)
    const float* W    = (const float*)d_in[1];   // (1536, 512)
    const float* bias = (const float*)d_in[2];   // (1536,)
    float* out = (float*)d_out;                  // [output (B,S,H) | c_last (B,H)]

    char* ws = (char*)d_ws;
    const size_t szAbf = (size_t)MM * DD * sizeof(unsigned short);   // 64 MiB
    const size_t szWbf = (size_t)N3 * DD * sizeof(unsigned short);   // 1.5 MiB
    const size_t szZ   = (size_t)MM * HH * sizeof(float);            // 128 MiB
    const size_t szC   = (size_t)NCH * CTOT * sizeof(float);         // 2 MiB

    unsigned short* Ahi = (unsigned short*)(ws);
    unsigned short* Alo = (unsigned short*)(ws + szAbf);
    unsigned short* Whi = (unsigned short*)(ws + 2 * szAbf);
    unsigned short* Wlo = (unsigned short*)(ws + 2 * szAbf + szWbf);
    float* zg    = (float*)(ws + 2 * szAbf + 2 * szWbf);
    float* fg    = (float*)(ws + 2 * szAbf + 2 * szWbf + szZ);
    float* cA    = (float*)(ws + 2 * szAbf + 2 * szWbf + 2 * szZ);
    float* cH    = (float*)(ws + 2 * szAbf + 2 * szWbf + 2 * szZ + szC);
    float* carry = (float*)(ws + 2 * szAbf + 2 * szWbf + 2 * szZ + 2 * szC);

    // 1) split-convert inp and W to bf16 hi/lo
    {
        const int n4 = MM * DD / 4;                        // 8388608
        qrnn_split<<<n4 / 256, 256, 0, stream>>>(inp, Ahi, Alo, n4);
    }
    {
        const int n4 = N3 * DD / 4;                        // 196608
        qrnn_split<<<n4 / 256, 256, 0, stream>>>(W, Whi, Wlo, n4);
    }

    // 2) WMMA GEMM + bias + activations
    {
        dim3 grid(N3 / 128, MM / 128);                     // (12, 512)
        qrnn_gemm_wmma<<<grid, 256, 0, stream>>>(Ahi, Alo, Whi, Wlo, bias,
                                                 zg, fg, out);
    }

    // 3) chunked linear-recurrence scan
    qrnn_scan_chunks<<<(CTOT * NCH) / 256, 256, 0, stream>>>(zg, fg, cA, cH);
    qrnn_scan_combine<<<CTOT / 256, 256, 0, stream>>>(cA, cH, carry,
                                                      out + (size_t)MM * HH);
    qrnn_scan_apply<<<(CTOT * NCH) / 256, 256, 0, stream>>>(zg, fg, carry, out);
}
